// Attention_41686952575399
// MI455X (gfx1250) — compile-verified, matrix operand order checked by reading, not run on hardware
//
#include <hip/hip_runtime.h>
#include <hip/hip_bf16.h>

typedef _Float16 half_t;
typedef __attribute__((ext_vector_type(8)))  _Float16 v8h;
typedef __attribute__((ext_vector_type(16))) _Float16 v16h;
typedef __attribute__((ext_vector_type(8)))  float    v8f;

#define B_   2
#define N_   1024
#define DIM_ 1024
#define H_   16
#define DH_  64
#define WIN_ 64
#define CB_  16
#define NB_  64
#define TOK_ (B_ * N_)

// ---------------------------------------------------------------- WMMA core
__device__ __forceinline__ v8f wmma16(v16h a, v16h b, v8f c) {
  return __builtin_amdgcn_wmma_f32_16x16x32_f16(false, a, false, b, (short)0, c, false, false);
}

__device__ __forceinline__ v16h cat8(v8h lo, v8h hi) {
  return __builtin_shufflevector(lo, hi, 0, 1, 2, 3, 4, 5, 6, 7, 8, 9, 10, 11, 12, 13, 14, 15);
}

// A-operand (16xK=32) from row-major [M=16, K] tile, row stride ld (halves).
// Also valid as B-operand when the source is row-major [N=16, K] (i.e. B^T).
// Lane L: row = L%16 ; halves 0..7 -> K = 8*(L/16)+t ; halves 8..15 -> +16.
// Two 16-byte loads per lane; requires 16B alignment (all strides are mult. of 8 halves).
__device__ __forceinline__ v16h ld_a_rm(const half_t* src, int ld, int lane) {
  const half_t* p = src + (size_t)(lane & 15) * ld + ((lane >> 4) << 3);
  return cat8(*(const v8h*)p, *(const v8h*)(p + 16));
}

// B^T-form loader with the K-run start clamped into [0, maxrun] (used for windowed V).
// Every 8-half run is entirely inside or entirely outside the window, and the
// outside columns of P are exactly zero, so clamped (garbage) runs contribute 0.
__device__ __forceinline__ v16h ld_bt_clamp(const half_t* base, int ld, int j0, int maxrun,
                                            int lane) {
  const int row = lane & 15;
  const int kb  = (lane >> 4) << 3;
  int c0 = j0 + kb, c1 = j0 + kb + 16;
  c0 = c0 < 0 ? 0 : (c0 > maxrun ? maxrun : c0);
  c1 = c1 < 0 ? 0 : (c1 > maxrun ? maxrun : c1);
  const half_t* p = base + (size_t)row * ld;
  return cat8(*(const v8h*)(p + c0), *(const v8h*)(p + c1));
}

__device__ __forceinline__ float redmax16(float v) {
  v = fmaxf(v, __shfl_xor(v, 1, 32));
  v = fmaxf(v, __shfl_xor(v, 2, 32));
  v = fmaxf(v, __shfl_xor(v, 4, 32));
  v = fmaxf(v, __shfl_xor(v, 8, 32));
  return v;
}
__device__ __forceinline__ float redsum16(float v) {
  v += __shfl_xor(v, 1, 32);
  v += __shfl_xor(v, 2, 32);
  v += __shfl_xor(v, 4, 32);
  v += __shfl_xor(v, 8, 32);
  return v;
}

// ---------------------------------------------------------------- prep kernels
__global__ void f32_to_f16_kernel(const float* __restrict__ src, half_t* __restrict__ dst, int n) {
  int i = blockIdx.x * 256 + threadIdx.x;
  if (i < n) dst[i] = (half_t)src[i];
}

// dst[N][K] (f16) = transpose of src[K][N] (f32); 32x32 LDS tiles, block (32,8).
__global__ void transpose_cvt_kernel(const float* __restrict__ src, half_t* __restrict__ dst,
                                     int K, int N) {
  __shared__ float tile[32][33];
  const int k0 = blockIdx.x * 32, n0 = blockIdx.y * 32;
  const int tx = threadIdx.x, ty = threadIdx.y;
  for (int r = ty; r < 32; r += 8)
    tile[r][tx] = src[(size_t)(k0 + r) * N + n0 + tx];
  __syncthreads();
  for (int r = ty; r < 32; r += 8)
    dst[(size_t)(n0 + r) * K + k0 + tx] = (half_t)tile[tx][r];
}

// src k_cw: [H][o=DH][i=DH][t=CB] ; dst: [H][o][t*DH + i]  (B^T form for conv GEMM)
__global__ void conv_w_rearrange_kernel(const float* __restrict__ src, half_t* __restrict__ dst) {
  int idx = blockIdx.x * 256 + threadIdx.x;
  if (idx >= H_ * DH_ * CB_ * DH_) return;
  int ho = idx >> 10;          // h*DH + o
  int kk = idx & 1023;         // t*DH + i
  int t = kk >> 6, i = kk & 63;
  dst[idx] = (half_t)src[((size_t)ho * DH_ + i) * CB_ + t];
}

// RMSNorm -> x16, and gate = sigmoid(x @ Ws + bs)[:, :2]
__global__ void rmsnorm_gate_kernel(const float* __restrict__ inp, const float* __restrict__ rms_w,
                                    const float* __restrict__ Ws, const float* __restrict__ bs,
                                    half_t* __restrict__ x16, float* __restrict__ gate) {
  __shared__ float red[256];
  const int tok = blockIdx.x;
  const int tid = threadIdx.x;
  const float* xin = inp + (size_t)tok * DIM_;
  float ss = 0.f;
  for (int i = tid; i < DIM_; i += 256) { float v = xin[i]; ss += v * v; }
  red[tid] = ss; __syncthreads();
  for (int s = 128; s > 0; s >>= 1) { if (tid < s) red[tid] += red[tid + s]; __syncthreads(); }
  const float rs = rsqrtf(red[0] / (float)DIM_ + 1.1920929e-07f);
  __syncthreads();
  float s0 = 0.f, s1 = 0.f;
  for (int i = tid; i < DIM_; i += 256) {
    float xv = xin[i] * rs * rms_w[i];
    x16[(size_t)tok * DIM_ + i] = (half_t)xv;
    s0 += xv * Ws[i * 3 + 0];
    s1 += xv * Ws[i * 3 + 1];
  }
  red[tid] = s0; __syncthreads();
  for (int s = 128; s > 0; s >>= 1) { if (tid < s) red[tid] += red[tid + s]; __syncthreads(); }
  float g0 = red[0]; __syncthreads();
  red[tid] = s1; __syncthreads();
  for (int s = 128; s > 0; s >>= 1) { if (tid < s) red[tid] += red[tid + s]; __syncthreads(); }
  float g1 = red[0];
  if (tid == 0) {
    gate[(size_t)tok * 2 + 0] = 1.f / (1.f + __expf(-(g0 + bs[0])));
    gate[(size_t)tok * 2 + 1] = 1.f / (1.f + __expf(-(g1 + bs[1])));
  }
}

// ---------------------------------------------------------------- QKV GEMM
// [TOK,1024] x Wqkv^T[3072][1024]; scatter to q/k/v in [b,h,n,dh] f16 layout,
// and additionally write V transposed as vT[b,h,dh,n] for the PV GEMMs.
__global__ void qkv_gemm_kernel(const half_t* __restrict__ x16, const half_t* __restrict__ wqT16,
                                half_t* __restrict__ qf, half_t* __restrict__ kf,
                                half_t* __restrict__ vf, half_t* __restrict__ vT) {
  const int lane = threadIdx.x;
  const int m0 = blockIdx.x * 16;
  const int c0 = blockIdx.y * 64;
  v8f acc[4] = {};
  for (int k0 = 0; k0 < DIM_; k0 += 32) {
    v16h a = ld_a_rm(x16 + (size_t)m0 * DIM_ + k0, DIM_, lane);
#pragma unroll
    for (int nt = 0; nt < 4; ++nt)
      acc[nt] = wmma16(a, ld_a_rm(wqT16 + (size_t)(c0 + nt * 16) * DIM_ + k0, DIM_, lane),
                       acc[nt]);
  }
  const int which = c0 >> 10;
  const int h = (c0 & 1023) >> 6;
  half_t* dst = which == 0 ? qf : (which == 1 ? kf : vf);
  const int mofs = (lane >> 4) << 3;
  const int col = lane & 15;
#pragma unroll
  for (int nt = 0; nt < 4; ++nt) {
    int d = nt * 16 + col;
#pragma unroll
    for (int r = 0; r < 8; ++r) {
      int row = m0 + r + mofs;
      int bb = row >> 10, n = row & 1023;
      half_t hv = (half_t)acc[nt][r];
      dst[(((size_t)bb * H_ + h) * N_ + n) * DH_ + d] = hv;
      if (which == 2) vT[(((size_t)bb * H_ + h) * DH_ + d) * N_ + n] = hv;
    }
  }
}

// ---------------------------------------------------------------- compress conv
// A[c, t*64+i] = kv[b,h,c*16+t,i] + pos[h,t,i]  ==  kvflat[c*1024+kk] + posflat[kk]
// out[c,o] = A @ W^T[h] + bias[h,o].  transpose_out: 0 -> O[c][o] (ck), 1 -> O[o][c] (cvT)
__global__ void conv_compress_kernel(const half_t* __restrict__ kvf, const half_t* __restrict__ pos16,
                                     const half_t* __restrict__ wT16, const float* __restrict__ bias,
                                     half_t* __restrict__ out16, int transpose_out) {
  const int lane = threadIdx.x;
  const int m0 = blockIdx.x * 16;
  const int h = blockIdx.y, b = blockIdx.z;
  const int K = CB_ * DH_;  // 1024
  const half_t* A = kvf + ((size_t)b * H_ + h) * N_ * DH_;
  const half_t* P = pos16 + (size_t)h * K;
  const half_t* W = wT16 + (size_t)h * DH_ * K;
  v8f acc[4] = {};
  const int row = lane & 15, kb = (lane >> 4) << 3;
  for (int k0 = 0; k0 < K; k0 += 32) {
    const half_t* ap = A + (size_t)(m0 + row) * K + k0 + kb;
    const half_t* pp = P + k0 + kb;
    v8h a0 = *(const v8h*)ap + *(const v8h*)pp;
    v8h a1 = *(const v8h*)(ap + 16) + *(const v8h*)(pp + 16);
    v16h a = cat8(a0, a1);
#pragma unroll
    for (int nt = 0; nt < 4; ++nt)
      acc[nt] = wmma16(a, ld_a_rm(W + (size_t)(nt * 16) * K + k0, K, lane), acc[nt]);
  }
  const int mofs = (lane >> 4) << 3;
  half_t* O = out16 + ((size_t)b * H_ + h) * NB_ * DH_;
#pragma unroll
  for (int nt = 0; nt < 4; ++nt) {
    int o = nt * 16 + (lane & 15);
    float bi = bias[h * DH_ + o];
#pragma unroll
    for (int r = 0; r < 8; ++r) {
      int c = m0 + r + mofs;
      half_t hv = (half_t)(acc[nt][r] + bi);
      if (transpose_out) O[(size_t)o * NB_ + c] = hv;
      else               O[(size_t)c * DH_ + o] = hv;
    }
  }
}

// ---------------------------------------------------------------- compressed attention
__global__ void compress_attn_kernel(const half_t* __restrict__ qf, const half_t* __restrict__ ck16,
                                     const half_t* __restrict__ cvT16, const float* __restrict__ gate,
                                     float* __restrict__ cout) {
  __shared__ alignas(16) half_t pl[16 * NB_];
  const int lane = threadIdx.x;
  const int m0 = blockIdx.x * 16;
  const int h = blockIdx.y, b = blockIdx.z;
  const half_t* Q   = qf + (((size_t)b * H_ + h) * N_ + m0) * DH_;
  const half_t* CK  = ck16 + ((size_t)b * H_ + h) * NB_ * DH_;   // [c][o]
  const half_t* CVT = cvT16 + ((size_t)b * H_ + h) * DH_ * NB_;  // [o][c]
  v8f s[4] = {};
  for (int k0 = 0; k0 < DH_; k0 += 32) {
    v16h a = ld_a_rm(Q + k0, DH_, lane);
#pragma unroll
    for (int jt = 0; jt < 4; ++jt)
      s[jt] = wmma16(a, ld_a_rm(CK + (size_t)(jt * 16) * DH_ + k0, DH_, lane), s[jt]);
  }
  const int mofs = (lane >> 4) << 3;
  const int col = lane & 15;
  float pv[4][8];
#pragma unroll
  for (int r = 0; r < 8; ++r) {
    int i = m0 + r + mofs;
    float tmp[4]; float mx = -1e30f;
#pragma unroll
    for (int jt = 0; jt < 4; ++jt) {
      int j = jt * 16 + col;
      float v = (j * CB_ + CB_ - 1 <= i) ? s[jt][r] * 0.125f : -1e30f;
      tmp[jt] = v; mx = fmaxf(mx, v);
    }
    mx = redmax16(mx);
    float l = 0.f;
#pragma unroll
    for (int jt = 0; jt < 4; ++jt) { tmp[jt] = __expf(tmp[jt] - mx); l += tmp[jt]; }
    l = redsum16(l);
    float inv = 1.f / l;
#pragma unroll
    for (int jt = 0; jt < 4; ++jt) pv[jt][r] = tmp[jt] * inv;
  }
#pragma unroll
  for (int jt = 0; jt < 4; ++jt)
#pragma unroll
    for (int r = 0; r < 8; ++r)
      pl[(r + mofs) * NB_ + jt * 16 + col] = (half_t)pv[jt][r];
  __syncthreads();
  v8f o[4] = {};
  for (int k0 = 0; k0 < NB_; k0 += 32) {
    v16h a = ld_a_rm(pl + k0, NB_, lane);
#pragma unroll
    for (int nt = 0; nt < 4; ++nt)
      o[nt] = wmma16(a, ld_a_rm(CVT + (size_t)(nt * 16) * NB_ + k0, NB_, lane), o[nt]);
  }
#pragma unroll
  for (int nt = 0; nt < 4; ++nt) {
    int d = nt * 16 + col;
#pragma unroll
    for (int r = 0; r < 8; ++r) {
      int i = m0 + r + mofs;
      float w0 = (i >= CB_ - 1) ? gate[((size_t)b * N_ + i) * 2 + 0] : 0.f;
      cout[((size_t)b * N_ + i) * DIM_ + h * DH_ + d] = o[nt][r] * w0;
    }
  }
}

// ---------------------------------------------------------------- sliding-window attention
// 16-query tile sees keys in [m0-64, m0+15] -> 5 key tiles; combine with cout via gates.
__global__ void local_attn_kernel(const half_t* __restrict__ qf, const half_t* __restrict__ kf,
                                  const half_t* __restrict__ vT, const float* __restrict__ gate,
                                  const float* __restrict__ cout, half_t* __restrict__ attn16) {
  __shared__ alignas(16) half_t pl[16 * 96];
  const int lane = threadIdx.x;
  const int m0 = blockIdx.x * 16;
  const int h = blockIdx.y, b = blockIdx.z;
  const half_t* Q  = qf + (((size_t)b * H_ + h) * N_ + m0) * DH_;
  const half_t* K  = kf + ((size_t)b * H_ + h) * N_ * DH_;
  const half_t* VT = vT + ((size_t)b * H_ + h) * DH_ * N_;  // [d][n]
  v8f s[5] = {};
  for (int k0 = 0; k0 < DH_; k0 += 32) {
    v16h a = ld_a_rm(Q + k0, DH_, lane);
#pragma unroll
    for (int t = 0; t < 5; ++t) {
      int j0 = m0 - 64 + t * 16;
      int jb = j0 < 0 ? 0 : j0;
      s[t] = wmma16(a, ld_a_rm(K + (size_t)jb * DH_ + k0, DH_, lane), s[t]);
    }
  }
  const int mofs = (lane >> 4) << 3;
  const int col = lane & 15;
  float pv[5][8];
#pragma unroll
  for (int r = 0; r < 8; ++r) {
    int i = m0 + r + mofs;
    float tmp[5]; float mx = -1e30f;
#pragma unroll
    for (int t = 0; t < 5; ++t) {
      int j = m0 - 64 + t * 16 + col;
      bool ok = (j >= 0) && (j <= i) && (i - j < WIN_);
      float v = ok ? s[t][r] * 0.125f : -1e30f;
      tmp[t] = v; mx = fmaxf(mx, v);
    }
    mx = redmax16(mx);
    float l = 0.f;
#pragma unroll
    for (int t = 0; t < 5; ++t) { tmp[t] = __expf(tmp[t] - mx); l += tmp[t]; }
    l = redsum16(l);
    float inv = 1.f / l;
#pragma unroll
    for (int t = 0; t < 5; ++t) pv[t][r] = tmp[t] * inv;
  }
#pragma unroll
  for (int t = 0; t < 5; ++t)
#pragma unroll
    for (int r = 0; r < 8; ++r)
      pl[(r + mofs) * 96 + t * 16 + col] = (half_t)pv[t][r];
#pragma unroll
  for (int r = 0; r < 8; ++r) pl[(r + mofs) * 96 + 80 + col] = (half_t)0.f;  // zero pad K=80..95
  __syncthreads();
  v8f o[4] = {};
  for (int ks = 0; ks < 3; ++ks) {
    v16h a = ld_a_rm(pl + ks * 32, 96, lane);
    int jbase = m0 - 64 + ks * 32;
#pragma unroll
    for (int nt = 0; nt < 4; ++nt)
      o[nt] = wmma16(a, ld_bt_clamp(VT + (size_t)(nt * 16) * N_, N_, jbase, N_ - 8, lane), o[nt]);
  }
#pragma unroll
  for (int nt = 0; nt < 4; ++nt) {
    int d = nt * 16 + col;
#pragma unroll
    for (int r = 0; r < 8; ++r) {
      int i = m0 + r + mofs;
      float w1 = gate[((size_t)b * N_ + i) * 2 + 1];
      size_t idx = ((size_t)b * N_ + i) * DIM_ + h * DH_ + d;
      attn16[idx] = (half_t)(o[nt][r] * w1 + cout[idx]);
    }
  }
}

// ---------------------------------------------------------------- output GEMM
__global__ void out_gemm_kernel(const half_t* __restrict__ attn16, const half_t* __restrict__ woutT16,
                                float* __restrict__ out) {
  const int lane = threadIdx.x;
  const int m0 = blockIdx.x * 16;
  const int c0 = blockIdx.y * 64;
  v8f acc[4] = {};
  for (int k0 = 0; k0 < DIM_; k0 += 32) {
    v16h a = ld_a_rm(attn16 + (size_t)m0 * DIM_ + k0, DIM_, lane);
#pragma unroll
    for (int nt = 0; nt < 4; ++nt)
      acc[nt] = wmma16(a, ld_a_rm(woutT16 + (size_t)(c0 + nt * 16) * DIM_ + k0, DIM_, lane),
                       acc[nt]);
  }
  const int mofs = (lane >> 4) << 3;
  const int col = lane & 15;
#pragma unroll
  for (int nt = 0; nt < 4; ++nt)
#pragma unroll
    for (int r = 0; r < 8; ++r)
      out[(size_t)(m0 + r + mofs) * DIM_ + c0 + nt * 16 + col] = acc[nt][r];
}

// ---------------------------------------------------------------- launch
extern "C" void kernel_launch(void* const* d_in, const int* in_sizes, int n_in,
                              void* d_out, int out_size, void* d_ws, size_t ws_size,
                              hipStream_t stream) {
  (void)in_sizes; (void)n_in; (void)out_size; (void)ws_size;
  const float* inp   = (const float*)d_in[0];
  const float* rms_w = (const float*)d_in[1];
  const float* Wqkv  = (const float*)d_in[2];
  const float* k_pos = (const float*)d_in[3];
  const float* v_pos = (const float*)d_in[4];
  const float* k_cw  = (const float*)d_in[5];
  const float* k_cb  = (const float*)d_in[6];
  const float* v_cw  = (const float*)d_in[7];
  const float* v_cb  = (const float*)d_in[8];
  const float* Ws    = (const float*)d_in[9];
  const float* bs    = (const float*)d_in[10];
  const float* Wout  = (const float*)d_in[11];
  float* out = (float*)d_out;

  char* wp = (char*)d_ws;
  auto alloc = [&](size_t bytes) -> void* {
    void* p = (void*)wp; wp += (bytes + 255) & ~(size_t)255; return p;
  };
  half_t* x16     = (half_t*)alloc((size_t)TOK_ * DIM_ * 2);
  float*  gate    = (float*) alloc((size_t)TOK_ * 2 * 4);
  half_t* wqkvT16 = (half_t*)alloc((size_t)DIM_ * 3 * DIM_ * 2);   // [3072][1024]
  half_t* woutT16 = (half_t*)alloc((size_t)DIM_ * DIM_ * 2);       // [1024][1024] transposed
  half_t* qf      = (half_t*)alloc((size_t)TOK_ * DIM_ * 2);
  half_t* kf      = (half_t*)alloc((size_t)TOK_ * DIM_ * 2);
  half_t* vf      = (half_t*)alloc((size_t)TOK_ * DIM_ * 2);
  half_t* vT      = (half_t*)alloc((size_t)TOK_ * DIM_ * 2);       // [b,h,d,n]
  half_t* kpos16  = (half_t*)alloc((size_t)H_ * CB_ * DH_ * 2);
  half_t* vpos16  = (half_t*)alloc((size_t)H_ * CB_ * DH_ * 2);
  half_t* wkT16   = (half_t*)alloc((size_t)H_ * DH_ * CB_ * DH_ * 2);
  half_t* wvT16   = (half_t*)alloc((size_t)H_ * DH_ * CB_ * DH_ * 2);
  half_t* ck16    = (half_t*)alloc((size_t)B_ * H_ * NB_ * DH_ * 2);
  half_t* cvT16   = (half_t*)alloc((size_t)B_ * H_ * DH_ * NB_ * 2);
  float*  cout    = (float*) alloc((size_t)TOK_ * DIM_ * 4);
  half_t* attn16  = (half_t*)alloc((size_t)TOK_ * DIM_ * 2);

  const int nPos = H_ * CB_ * DH_, nCw = H_ * DH_ * CB_ * DH_;
  transpose_cvt_kernel<<<dim3(DIM_ / 32, 3 * DIM_ / 32), dim3(32, 8), 0, stream>>>(
      Wqkv, wqkvT16, DIM_, 3 * DIM_);
  transpose_cvt_kernel<<<dim3(DIM_ / 32, DIM_ / 32), dim3(32, 8), 0, stream>>>(
      Wout, woutT16, DIM_, DIM_);
  f32_to_f16_kernel<<<(nPos + 255) / 256, 256, 0, stream>>>(k_pos, kpos16, nPos);
  f32_to_f16_kernel<<<(nPos + 255) / 256, 256, 0, stream>>>(v_pos, vpos16, nPos);
  conv_w_rearrange_kernel<<<(nCw + 255) / 256, 256, 0, stream>>>(k_cw, wkT16);
  conv_w_rearrange_kernel<<<(nCw + 255) / 256, 256, 0, stream>>>(v_cw, wvT16);

  rmsnorm_gate_kernel<<<TOK_, 256, 0, stream>>>(inp, rms_w, Ws, bs, x16, gate);
  qkv_gemm_kernel<<<dim3(TOK_ / 16, 3 * DIM_ / 64), 32, 0, stream>>>(x16, wqkvT16, qf, kf, vf, vT);
  conv_compress_kernel<<<dim3(NB_ / 16, H_, B_), 32, 0, stream>>>(kf, kpos16, wkT16, k_cb, ck16, 0);
  conv_compress_kernel<<<dim3(NB_ / 16, H_, B_), 32, 0, stream>>>(vf, vpos16, wvT16, v_cb, cvT16, 1);
  compress_attn_kernel<<<dim3(N_ / 16, H_, B_), 32, 0, stream>>>(qf, ck16, cvT16, gate, cout);
  local_attn_kernel<<<dim3(N_ / 16, H_, B_), 32, 0, stream>>>(qf, kf, vT, gate, cout, attn16);
  out_gemm_kernel<<<dim3(TOK_ / 16, DIM_ / 64), 32, 0, stream>>>(attn16, woutT16, out);
}
